// TreeDecoder_24927990186148
// MI455X (gfx1250) — compile-verified, matrix operand order checked by reading, not run on hardware
//
#include <hip/hip_runtime.h>

// ---------------------------------------------------------------------------
// TreeDecoder on MI455X (gfx1250).
// Key insight: repeat(h, degree) + per-tree enc_x means all nodes at a level
// of a tree are identical -> only 11 distinct hidden rows per tree.
// Compute is ~0.37 GFLOP; the 268MB broadcast store dominates (~11.5us @ 23.3TB/s).
// ---------------------------------------------------------------------------

typedef __bf16 bf16_t;
typedef __attribute__((ext_vector_type(16))) __bf16 v16bf;
typedef __attribute__((ext_vector_type(8)))  float  v8f;
typedef __attribute__((ext_vector_type(4)))  float  f32x4;
typedef __attribute__((ext_vector_type(2)))  float  f32x2;

#define HS    256      // hidden size == num classes
#define BB    128      // batch (trees)
#define NLEV  11       // depth+1 distinct levels
#define NODES 2047     // 2^(DEPTH+1)-1

__device__ __forceinline__ void bf16_split(float x, bf16_t &hi, bf16_t &lo) {
    hi = (bf16_t)x;
    lo = (bf16_t)(x - (float)hi);
}

// Accumulate A[16 x 256] * B[256 x 16] into acc (fp32), using bf16 WMMA with
// a 3-term hi/lo split for ~fp32 accuracy.  A,B row-major fp32.
// Register layouts per CDNA5 ISA 7.12.2:
//   A 16x32 bf16 : lanes 0-15 rows 0-15 (lanes 16-31 same rows, K+8 group),
//                  VGPR v holds K-pair {base+2v, base+2v+1}.
//   B 32x16 bf16 : lane holds column (lane&15), same K grouping.
//   C/D 16x16 f32: VGPR r = row r + 8*(lane>=16), col = lane&15.
__device__ __forceinline__ v8f wmma_tile_k256(const float* __restrict__ A, int lda,
                                              const float* __restrict__ B, int ldb,
                                              v8f acc) {
    const int lane = threadIdx.x & 31;
    const int half = (lane >> 4) & 1;
    const int arow = lane & 15;
    const int bcol = lane & 15;

    for (int k0 = 0; k0 < HS; k0 += 32) {
        v16bf a_hi, a_lo, b_hi, b_lo;
#pragma unroll
        for (int v = 0; v < 8; ++v) {
            const int kb = ((v < 4) ? 0 : 16) + 8 * half + 2 * (v & 3);
            const int k  = k0 + kb;
            bf16_t h0, l0, h1, l1;
            // A: two consecutive K of this lane's row
            f32x2 af = *(const f32x2*)(A + arow * lda + k);
            bf16_split(af.x, h0, l0);
            bf16_split(af.y, h1, l1);
            a_hi[2*v] = h0;  a_hi[2*v+1] = h1;
            a_lo[2*v] = l0;  a_lo[2*v+1] = l1;
            // B: two consecutive K rows of this lane's column
            bf16_split(B[(k    ) * ldb + bcol], h0, l0);
            bf16_split(B[(k + 1) * ldb + bcol], h1, l1);
            b_hi[2*v] = h0;  b_hi[2*v+1] = h1;
            b_lo[2*v] = l0;  b_lo[2*v+1] = l1;
        }
        acc = __builtin_amdgcn_wmma_f32_16x16x32_bf16(false, a_hi, false, b_hi, (short)0, acc, false, false);
        acc = __builtin_amdgcn_wmma_f32_16x16x32_bf16(false, a_hi, false, b_lo, (short)0, acc, false, false);
        acc = __builtin_amdgcn_wmma_f32_16x16x32_bf16(false, a_lo, false, b_hi, (short)0, acc, false, false);
    }
    return acc;
}

// Device-wide sense barrier (one counter per use; counters pre-zeroed by memset).
__device__ __forceinline__ void gbar(unsigned* cnt, unsigned nblocks, int tid) {
    __syncthreads();
    if (tid == 0) {
        __hip_atomic_fetch_add(cnt, 1u, __ATOMIC_RELEASE, __HIP_MEMORY_SCOPE_AGENT);
        while (__hip_atomic_load(cnt, __ATOMIC_ACQUIRE, __HIP_MEMORY_SCOPE_AGENT) < nblocks) {
            __builtin_amdgcn_s_sleep(1);
        }
    }
    __syncthreads();
}

// K1: enc_x = encs@W_enc + b ; h_0 = tanh(enc_x) ; h_l = tanh(h_{l-1}@U + enc_x)
// Grid: 8 blocks x 512 threads. Block j owns output columns [32j, 32j+32).
// Wave w: m-tile = w>>1, n-tile = w&1.  Each wave keeps its enc_x tile in VGPRs.
__global__ void __launch_bounds__(512)
k_recurrence(const float* __restrict__ encs, const float* __restrict__ W_enc,
             const float* __restrict__ U,    const float* __restrict__ bvec,
             float* __restrict__ Hsmall, unsigned* barcnt, unsigned nblocks) {
    const int tid  = threadIdx.x;
    const int wave = tid >> 5;
    const int lane = tid & 31;
    const int half = (lane >> 4) & 1;
    const int m0   = (wave >> 1) * 16;
    const int n0   = blockIdx.x * 32 + (wave & 1) * 16;
    const int col  = n0 + (lane & 15);

    // ---- enc_x tile (kept in registers for the whole recurrence) ----
    v8f enc = {0.f,0.f,0.f,0.f,0.f,0.f,0.f,0.f};
    enc = wmma_tile_k256(encs + m0 * HS, HS, W_enc + n0, HS, enc);
    const float bias = bvec[col];
#pragma unroll
    for (int r = 0; r < 8; ++r) enc[r] += bias;

    // ---- level 0 ----
#pragma unroll
    for (int r = 0; r < 8; ++r) {
        const int row = m0 + r + 8 * half;
        Hsmall[row * HS + col] = tanhf(enc[r]);
    }
    gbar(barcnt + 0, nblocks, tid);

    // ---- levels 1..10 (serial recurrence, parallel within level) ----
    for (int l = 1; l < NLEV; ++l) {
        v8f acc = {0.f,0.f,0.f,0.f,0.f,0.f,0.f,0.f};
        acc = wmma_tile_k256(Hsmall + (l - 1) * BB * HS + m0 * HS, HS, U + n0, HS, acc);
#pragma unroll
        for (int r = 0; r < 8; ++r) {
            const int row = m0 + r + 8 * half;
            Hsmall[l * BB * HS + row * HS + col] = tanhf(acc[r] + enc[r]);
        }
        gbar(barcnt + l, nblocks, tid);
    }
}

// K2: Ltab[1408 x 256] = Hsmall[1408 x 256] @ W_out + b_out.
// Grid: 88 blocks x 512 threads; block = one 16-row m-tile, wave = one n-tile.
__global__ void __launch_bounds__(512)
k_logits(const float* __restrict__ Hsmall, const float* __restrict__ W_out,
         const float* __restrict__ b_out, float* __restrict__ Ltab) {
    const int tid  = threadIdx.x;
    const int wave = tid >> 5;
    const int lane = tid & 31;
    const int half = (lane >> 4) & 1;
    const int m0   = blockIdx.x * 16;
    const int n0   = wave * 16;
    const int col  = n0 + (lane & 15);

    v8f acc = {0.f,0.f,0.f,0.f,0.f,0.f,0.f,0.f};
    acc = wmma_tile_k256(Hsmall + m0 * HS, HS, W_out + n0, HS, acc);
    const float bias = b_out[col];
#pragma unroll
    for (int r = 0; r < 8; ++r) {
        const int row = m0 + r + 8 * half;
        Ltab[row * HS + col] = acc[r] + bias;
    }
}

// K3: broadcast 1.4MB logit table into the 268MB output.
// out[b][node][c] = Ltab[level(node)*128 + b][c], level = floor(log2(node+1)).
// Pure store-bandwidth phase: float4 + nontemporal stores.
__global__ void __launch_bounds__(256)
k_broadcast(const float* __restrict__ Ltab, float* __restrict__ out) {
    const long long total = (long long)BB * NODES * (HS / 4);
    const f32x4* __restrict__ L4 = (const f32x4*)Ltab;
    f32x4* __restrict__ O4 = (f32x4*)out;
    const long long stride = (long long)gridDim.x * blockDim.x;
    for (long long f = (long long)blockIdx.x * blockDim.x + threadIdx.x;
         f < total; f += stride) {
        const int row  = (int)(f >> 6);          // (b, node) pair; 64 float4 per row
        const int c4   = (int)(f & 63);
        const int b    = row / NODES;
        const int node = row - b * NODES;
        const int lvl  = 31 - __clz(node + 1);
        const f32x4 v  = L4[(lvl * BB + b) * (HS / 4) + c4];
        __builtin_nontemporal_store(v, &O4[f]);
    }
}

extern "C" void kernel_launch(void* const* d_in, const int* in_sizes, int n_in,
                              void* d_out, int out_size, void* d_ws, size_t ws_size,
                              hipStream_t stream) {
    const float* encs  = (const float*)d_in[0];
    const float* W_enc = (const float*)d_in[1];
    const float* U     = (const float*)d_in[2];
    const float* bvec  = (const float*)d_in[3];
    const float* W_out = (const float*)d_in[4];
    const float* b_out = (const float*)d_in[5];
    float* out = (float*)d_out;

    // Workspace layout: Hsmall [11*128*256] | Ltab [11*128*256] | barrier counters
    float*    Hsmall = (float*)d_ws;
    float*    Ltab   = Hsmall + (size_t)NLEV * BB * HS;
    unsigned* barcnt = (unsigned*)(Ltab + (size_t)NLEV * BB * HS);

    hipMemsetAsync(barcnt, 0, 16 * sizeof(unsigned), stream);

    const unsigned nblk = 8;
    k_recurrence<<<nblk, 512, 0, stream>>>(encs, W_enc, U, bvec, Hsmall, barcnt, nblk);
    k_logits<<<(NLEV * BB) / 16, 512, 0, stream>>>(Hsmall, W_out, b_out, Ltab);
    k_broadcast<<<8192, 256, 0, stream>>>(Ltab, out);
}
